// GRUNet_22986664969033
// MI455X (gfx1250) — matrix operand order repaired and checked by reading, not compiled, not run on hardware
//
#include <hip/hip_runtime.h>

// ---------------- problem constants ----------------
#define IN_DIM   10
#define HID      100
#define G3       300            // 3*HID gate rows
#define G3P      304            // padded gate rows (19 tiles * 16)
#define OUT_DIM  3
#define BATCH    512
#define SEQ      512
#define NT       19             // N tiles of 16 gate columns
#define THREADS  (NT * 32)      // 608 threads = 19 wave32s, one wave per tile
#define CSTR     20             // column-major gate buffer stride (f32), 80B rows
#define HBSTR    136            // hidden bf16 staging row stride (16B aligned rows)
#define XTSTR    40             // x_t bf16 staging row stride

// padded bf16 weight layout inside d_ws (element offsets)
#define WIH0_OFF 0
#define WHH0_OFF (G3P * 32)
#define WIH1_OFF (WHH0_OFF + G3P * 128)
#define WHH1_OFF (WIH1_OFF + G3P * 128)
#define WS_ELEMS (WHH1_OFF + G3P * 128)

typedef __attribute__((ext_vector_type(16))) __bf16       v16bf;
typedef __attribute__((ext_vector_type(8)))  float        v8f;
typedef __attribute__((ext_vector_type(4)))  float        f32x4;
typedef __attribute__((ext_vector_type(4)))  unsigned int u32x4;

// float -> bf16 (round to nearest even)
static __device__ __forceinline__ unsigned short f2bf_bits(float f) {
  unsigned u = __builtin_bit_cast(unsigned, f);
  u += 0x7FFFu + ((u >> 16) & 1u);
  return (unsigned short)(u >> 16);
}
static __device__ __forceinline__ __bf16 f2bf(float f) {
  unsigned short h = f2bf_bits(f);
  return __builtin_bit_cast(__bf16, h);
}
static __device__ __forceinline__ float sigm(float x) {
  return 1.0f / (1.0f + __expf(-x));
}

// A fragment (16x32 bf16) from an LDS row-major [16][stride] tile.
// ISA 16-bit A layout: lane L holds row m=L&15; element v holds
// k = kbase + (v&7) + 8*(L>>4) + 16*(v>>3)  -> two contiguous 16B chunks.
static __device__ __forceinline__ v16bf load_a(const __bf16* base, int stride,
                                               int lane, int kbase) {
  const int m = lane & 15, kh = lane >> 4;
  const __bf16* p = base + m * stride + kbase + kh * 8;
  union { u32x4 q[2]; v16bf v; } u;
  u.q[0] = *(const u32x4*)(const void*)p;         // k = kbase+8*kh    .. +7
  u.q[1] = *(const u32x4*)(const void*)(p + 16);  // k = kbase+16+8*kh .. +7
  return u.v;
}

// B fragment (32x16 bf16) from global padded row-major weights [304][kpad]
// (row = gate output column n). ISA 16-bit B layout: lane L holds column
// n=L&15, k = kbase + (v&7) + 8*(L>>4) + 16*(v>>3)  -> two 16B chunks.
static __device__ __forceinline__ v16bf load_b(const __bf16* w, int kpad,
                                               int row0, int lane, int kbase) {
  const int n = lane & 15, kh = lane >> 4;
  const __bf16* p = w + (size_t)(row0 + n) * kpad + kbase + kh * 8;
  union { u32x4 q[2]; v16bf v; } u;
  u.q[0] = *(const u32x4*)(const void*)p;
  u.q[1] = *(const u32x4*)(const void*)(p + 16);
  return u.v;
}

static __device__ __forceinline__ v8f wmma_bf16(v16bf a, v16bf b, v8f c) {
  // D = A(16x32) * B(32x16) + C, f32 accumulate
  return __builtin_amdgcn_wmma_f32_16x16x32_bf16(false, a, false, b,
                                                 (short)0, c, false, false);
}

static __device__ __forceinline__ void store4(float* p, float a, float b,
                                              float c, float d) {
  f32x4 v = {a, b, c, d};
  *(f32x4*)(void*)p = v;
}

// ---------------- weight pad/convert pre-pass ----------------
__global__ void pad_weights_kernel(const float* __restrict__ wih0,
                                   const float* __restrict__ whh0,
                                   const float* __restrict__ wih1,
                                   const float* __restrict__ whh1,
                                   unsigned short* __restrict__ ws) {
  int tid = blockIdx.x * blockDim.x + threadIdx.x;
  if (tid >= WS_ELEMS) return;
  float v = 0.0f;
  if (tid < WHH0_OFF) {                      // w_ih0 -> [304][32], K=10
    int r = tid >> 5, c = tid & 31;
    if (r < G3 && c < IN_DIM) v = wih0[r * IN_DIM + c];
  } else if (tid < WIH1_OFF) {               // w_hh0 -> [304][128], K=100
    int l = tid - WHH0_OFF, r = l >> 7, c = l & 127;
    if (r < G3 && c < HID) v = whh0[r * HID + c];
  } else if (tid < WHH1_OFF) {               // w_ih1 -> [304][128], K=100
    int l = tid - WIH1_OFF, r = l >> 7, c = l & 127;
    if (r < G3 && c < HID) v = wih1[r * HID + c];
  } else {                                   // w_hh1 -> [304][128], K=100
    int l = tid - WHH1_OFF, r = l >> 7, c = l & 127;
    if (r < G3 && c < HID) v = whh1[r * HID + c];
  }
  ws[tid] = f2bf_bits(v);
}

// ---------------- fused 2-layer GRU + head ----------------
__global__ __launch_bounds__(THREADS, 1)
void gru2_fused_kernel(const float* __restrict__ x,
                       const float* __restrict__ h_in,
                       const float* __restrict__ bi0, const float* __restrict__ bh0,
                       const float* __restrict__ bi1, const float* __restrict__ bh1,
                       const float* __restrict__ fcw, const float* __restrict__ fcb,
                       const unsigned short* __restrict__ ws_u16,
                       float* __restrict__ out) {
  __shared__ __align__(16) __bf16 XT[16 * XTSTR];      // x_t staging (bf16, zero-pad K)
  __shared__ __align__(16) __bf16 HBF[2][16 * HBSTR];  // h bf16 GEMM operand, K padded
  __shared__ float HF[2][16 * HID];                    // h master copy (f32)
  __shared__ __align__(16) float GXc[G3P * CSTR];      // gates col-major: r,z sum; n x-side
  __shared__ __align__(16) float GHc[G3P * CSTR];      // gates col-major: n h-side
  __shared__ float BI[2][G3P];
  __shared__ float BH[2][G3P];

  const int tid  = threadIdx.x;
  const int lane = tid & 31;
  const int wave = tid >> 5;                 // == N-tile index, 0..18
  const int b0   = blockIdx.x * 16;

  const __bf16* ws   = reinterpret_cast<const __bf16*>(ws_u16);
  const __bf16* WIH0 = ws + WIH0_OFF;
  const __bf16* WHH0 = ws + WHH0_OFF;
  const __bf16* WIH1 = ws + WIH1_OFF;
  const __bf16* WHH1 = ws + WHH1_OFF;

  // ---- init LDS ----
  for (int e = tid; e < 16 * XTSTR; e += THREADS) XT[e] = f2bf(0.0f);
  for (int e = tid; e < 2 * 16 * HBSTR; e += THREADS) {
    int l = e / (16 * HBSTR);
    int r = e - l * (16 * HBSTR);
    int m = r / HBSTR;
    int j = r - m * HBSTR;
    float v = 0.0f;
    if (j < HID) {
      v = h_in[(size_t)(l * BATCH + b0 + m) * HID + j];
      HF[l][m * HID + j] = v;
    }
    HBF[l][m * HBSTR + j] = f2bf(v);
  }
  for (int e = tid; e < G3P; e += THREADS) {  // zero-padded biases: no guards later
    BI[0][e] = (e < G3) ? bi0[e] : 0.0f;
    BH[0][e] = (e < G3) ? bh0[e] : 0.0f;
    BI[1][e] = (e < G3) ? bi1[e] : 0.0f;
    BH[1][e] = (e < G3) ? bh1[e] : 0.0f;
  }

  // ---- hoist this wave's B (weight) fragments into registers for all 512 steps
  const int row0 = wave * 16;
  v16bf b0ih, b0hh[4], b1ih[4], b1hh[4];
  b0ih = load_b(WIH0, 32, row0, lane, 0);
#pragma unroll
  for (int ks = 0; ks < 4; ++ks) b0hh[ks] = load_b(WHH0, 128, row0, lane, ks * 32);
#pragma unroll
  for (int ks = 0; ks < 4; ++ks) b1ih[ks] = load_b(WIH1, 128, row0, lane, ks * 32);
#pragma unroll
  for (int ks = 0; ks < 4; ++ks) b1hh[ks] = load_b(WHH1, 128, row0, lane, ks * 32);

  const int col = row0 + (lane & 15);        // gate column this lane writes
  const int mb  = (lane >> 4) * 8;           // first of 8 batch rows in acc
  __syncthreads();

  for (int t = 0; t < SEQ; ++t) {
    // stage x_t as bf16 (cols 0..9; 10..31 stay zero)
    if (tid < 16 * IN_DIM) {
      int m = tid / IN_DIM, c = tid - m * IN_DIM;
      XT[m * XTSTR + c] = f2bf(x[((size_t)(b0 + m) * SEQ + t) * IN_DIM + c]);
    }
    if (tid < 16 && t + 1 < SEQ) {  // global_prefetch next x rows
      __builtin_prefetch(&x[((size_t)(b0 + tid) * SEQ + (t + 1)) * IN_DIM], 0, 3);
    }
    __syncthreads();

    // ---- layer 0: gates = x_t @ w_ih0^T + h0 @ w_hh0^T (+biases) ----
    {
      v8f accx = {};
      v8f acch = {};
      accx = wmma_bf16(load_a(XT, XTSTR, lane, 0), b0ih, accx);
#pragma unroll
      for (int ks = 0; ks < 4; ++ks)
        acch = wmma_bf16(load_a(HBF[0], HBSTR, lane, ks * 32), b0hh[ks], acch);
      const float bi = BI[0][col], bh = BH[0][col];
      float* gx = &GXc[col * CSTR + mb];
      if (col < 2 * HID) {                   // r,z: sum both sides
        store4(gx,     accx[0]+acch[0]+bi+bh, accx[1]+acch[1]+bi+bh,
                       accx[2]+acch[2]+bi+bh, accx[3]+acch[3]+bi+bh);
        store4(gx + 4, accx[4]+acch[4]+bi+bh, accx[5]+acch[5]+bi+bh,
                       accx[6]+acch[6]+bi+bh, accx[7]+acch[7]+bi+bh);
      } else {                               // n gate: keep sides separate
        float* gh = &GHc[col * CSTR + mb];
        store4(gx,     accx[0]+bi, accx[1]+bi, accx[2]+bi, accx[3]+bi);
        store4(gx + 4, accx[4]+bi, accx[5]+bi, accx[6]+bi, accx[7]+bi);
        store4(gh,     acch[0]+bh, acch[1]+bh, acch[2]+bh, acch[3]+bh);
        store4(gh + 4, acch[4]+bh, acch[5]+bh, acch[6]+bh, acch[7]+bh);
      }
    }
    __syncthreads();
    // ---- layer 0 pointwise GRU update ----
    for (int e = tid; e < 16 * HID; e += THREADS) {
      int m = e & 15, j = e >> 4;
      float r  = sigm(GXc[j * CSTR + m]);
      float z  = sigm(GXc[(HID + j) * CSTR + m]);
      float nn = tanhf(GXc[(2 * HID + j) * CSTR + m] +
                       r * GHc[(2 * HID + j) * CSTR + m]);
      float hn = (1.0f - z) * nn + z * HF[0][m * HID + j];
      HF[0][m * HID + j] = hn;
      HBF[0][m * HBSTR + j] = f2bf(hn);
    }
    __syncthreads();

    // ---- layer 1: gates = h0_new @ w_ih1^T + h1 @ w_hh1^T (+biases) ----
    {
      v8f accx = {};
      v8f acch = {};
#pragma unroll
      for (int ks = 0; ks < 4; ++ks) {
        accx = wmma_bf16(load_a(HBF[0], HBSTR, lane, ks * 32), b1ih[ks], accx);
        acch = wmma_bf16(load_a(HBF[1], HBSTR, lane, ks * 32), b1hh[ks], acch);
      }
      const float bi = BI[1][col], bh = BH[1][col];
      float* gx = &GXc[col * CSTR + mb];
      if (col < 2 * HID) {
        store4(gx,     accx[0]+acch[0]+bi+bh, accx[1]+acch[1]+bi+bh,
                       accx[2]+acch[2]+bi+bh, accx[3]+acch[3]+bi+bh);
        store4(gx + 4, accx[4]+acch[4]+bi+bh, accx[5]+acch[5]+bi+bh,
                       accx[6]+acch[6]+bi+bh, accx[7]+acch[7]+bi+bh);
      } else {
        float* gh = &GHc[col * CSTR + mb];
        store4(gx,     accx[0]+bi, accx[1]+bi, accx[2]+bi, accx[3]+bi);
        store4(gx + 4, accx[4]+bi, accx[5]+bi, accx[6]+bi, accx[7]+bi);
        store4(gh,     acch[0]+bh, acch[1]+bh, acch[2]+bh, acch[3]+bh);
        store4(gh + 4, acch[4]+bh, acch[5]+bh, acch[6]+bh, acch[7]+bh);
      }
    }
    __syncthreads();
    // ---- layer 1 pointwise GRU update ----
    for (int e = tid; e < 16 * HID; e += THREADS) {
      int m = e & 15, j = e >> 4;
      float r  = sigm(GXc[j * CSTR + m]);
      float z  = sigm(GXc[(HID + j) * CSTR + m]);
      float nn = tanhf(GXc[(2 * HID + j) * CSTR + m] +
                       r * GHc[(2 * HID + j) * CSTR + m]);
      float hn = (1.0f - z) * nn + z * HF[1][m * HID + j];
      HF[1][m * HID + j] = hn;
      HBF[1][m * HBSTR + j] = f2bf(hn);
    }
    __syncthreads();
  }

  // ---- epilogue: h_out then fc(relu(h1_last)) ----
  for (int e = tid; e < 2 * 16 * HID; e += THREADS) {
    int l = e / (16 * HID);
    int r = e - l * (16 * HID);
    int m = r / HID, j = r - m * HID;
    out[(size_t)BATCH * OUT_DIM + (size_t)(l * BATCH + b0 + m) * HID + j] = HF[l][r];
  }
  if (tid < 16 * OUT_DIM) {
    int m = tid / OUT_DIM, o = tid - m * OUT_DIM;
    float s = fcb[o];
    for (int j = 0; j < HID; ++j) {
      float v = HF[1][m * HID + j];
      v = v > 0.0f ? v : 0.0f;
      s += v * fcw[o * HID + j];
    }
    out[(size_t)(b0 + m) * OUT_DIM + o] = s;
  }
}

// ---------------- host entry ----------------
extern "C" void kernel_launch(void* const* d_in, const int* in_sizes, int n_in,
                              void* d_out, int out_size, void* d_ws, size_t ws_size,
                              hipStream_t stream) {
  (void)in_sizes; (void)n_in; (void)out_size; (void)ws_size;
  const float* x    = (const float*)d_in[0];
  const float* h    = (const float*)d_in[1];
  const float* wih0 = (const float*)d_in[2];
  const float* whh0 = (const float*)d_in[3];
  const float* bi0  = (const float*)d_in[4];
  const float* bh0  = (const float*)d_in[5];
  const float* wih1 = (const float*)d_in[6];
  const float* whh1 = (const float*)d_in[7];
  const float* bi1  = (const float*)d_in[8];
  const float* bh1  = (const float*)d_in[9];
  const float* fcw  = (const float*)d_in[10];
  const float* fcb  = (const float*)d_in[11];
  unsigned short* ws = (unsigned short*)d_ws;   // 252,928 bytes padded bf16 weights
  float* out = (float*)d_out;

  pad_weights_kernel<<<(WS_ELEMS + 255) / 256, 256, 0, stream>>>(wih0, whh0, wih1, whh1, ws);
  gru2_fused_kernel<<<BATCH / 16, THREADS, 0, stream>>>(x, h, bi0, bh0, bi1, bh1,
                                                        fcw, fcb, ws, out);
}
